// GNN_30227979829998
// MI455X (gfx1250) — compile-verified, matrix operand order checked by reading, not run on hardware
//
#include <hip/hip_runtime.h>

#define N_NODES 50000
#define N_EDGES 800000
#define N_GRAPHS 256
#define M_PAD 50048   // 391 * 128

typedef __attribute__((ext_vector_type(16))) __bf16 v16bf;
typedef __attribute__((ext_vector_type(8)))  float  v8f;

struct Frag32 { uint4 lo, hi; };

__device__ __forceinline__ unsigned short f2bf(float f) {
  unsigned int u = __float_as_uint(f);
  u += 0x7FFFu + ((u >> 16) & 1u);   // round-to-nearest-even
  return (unsigned short)(u >> 16);
}

__device__ __forceinline__ int lower_bound_i(const int* __restrict__ a, int n, int key) {
  int lo = 0, hi = n;
  while (lo < hi) { int mid = (lo + hi) >> 1; if (a[mid] < key) lo = mid + 1; else hi = mid; }
  return lo;
}

// ---------------- CSR build from sorted dst + inverse degree ----------------
__global__ void build_rowptr(const int* __restrict__ dst, int* __restrict__ row_ptr,
                             float* __restrict__ inv_deg) {
  int n = blockIdx.x * blockDim.x + threadIdx.x;
  if (n > N_NODES) return;
  int s = lower_bound_i(dst, N_EDGES, n);
  row_ptr[n] = s;
  if (n < N_NODES) {
    int e = lower_bound_i(dst, N_EDGES, n + 1);
    int deg = e - s;
    inv_deg[n] = deg > 0 ? 1.0f / (float)deg : 0.0f;
  }
}

// ---------------- mean aggregation (scatter-free, dst sorted) ----------------
__global__ __launch_bounds__(256) void aggregate(const float* __restrict__ x,
                                                 const int* __restrict__ src,
                                                 const int* __restrict__ row_ptr,
                                                 const float* __restrict__ inv_deg,
                                                 float* __restrict__ out) {
  int node = blockIdx.x * 4 + (threadIdx.x >> 6);
  int c = (threadIdx.x & 63) * 4;
  if (node >= N_NODES) return;
  int s = row_ptr[node], e = row_ptr[node + 1];
  float4 acc = {0.f, 0.f, 0.f, 0.f};
  for (int i = s; i < e; i++) {
    unsigned sn = (unsigned)src[i];
    float4 v = *(const float4*)(x + sn * 256u + c);
    acc.x += v.x; acc.y += v.y; acc.z += v.z; acc.w += v.w;
  }
  float id = inv_deg[node];
  acc.x *= id; acc.y *= id; acc.z *= id; acc.w *= id;
  *(float4*)(out + (unsigned)node * 256u + c) = acc;
}

// ---------------- weight transpose + fp32->bf16 convert ----------------
__global__ void wconv(const float* __restrict__ W, unsigned short* __restrict__ out,
                      int K, int N, int outPitch, int colOff) {
  int idx = blockIdx.x * 256 + threadIdx.x;
  if (idx >= K * N) return;
  int k = idx / N, n = idx % N;
  out[(unsigned)n * outPitch + colOff + k] = f2bf(W[idx]);
}

// ---------------- fused dual-input bf16 WMMA GEMM ----------------
// out[M][N] = [A0 | A1] (K-concat, each pitch 256 fp32) @ Wt^T + bias, opt. relu
// Wt: [N][Ktot] bf16 pre-transposed. Block tile 128x128, 8 waves (2Mx4N),
// wave tile 64x32 = 4x2 WMMA frags. Stores unconditional: out rows padded.
#define TILE_M 128
#define LDS_PITCH 40

__global__ __launch_bounds__(256)
void gemm_dual_bf16(const float* __restrict__ A0, const float* __restrict__ A1,
                    const unsigned short* __restrict__ Wt, const float* __restrict__ bias,
                    float* __restrict__ out, int M, int N, int Ktot, int doRelu) {
  __shared__ unsigned short As[TILE_M * LDS_PITCH];
  const int K0   = Ktot >> 1;
  const int tid  = threadIdx.x;
  const int wave = tid >> 5, lane = tid & 31;
  const int half = lane >> 4, l16 = lane & 15;
  const int wm = wave & 1, wn = wave >> 1;     // 2 (M) x 4 (N) waves
  const int mLoc  = wm * 64;                    // wave row offset within tile
  const int mBase = blockIdx.x * TILE_M + mLoc;
  const int nBase = blockIdx.y * 128 + wn * 32;

  const v8f vzero = {0.f, 0.f, 0.f, 0.f, 0.f, 0.f, 0.f, 0.f};
  v8f acc[4][2];
#pragma unroll
  for (int i = 0; i < 4; i++) { acc[i][0] = vzero; acc[i][1] = vzero; }

  for (int k0 = 0; k0 < Ktot; k0 += 32) {
    const float* base = (k0 < K0) ? A0 : A1;
    const int kk = (k0 < K0) ? k0 : (k0 - K0);
    __syncthreads();
#pragma unroll
    for (int i = 0; i < 4; i++) {
      int idx = tid + 256 * i;            // 1024 float4 slots = 128 rows x 8
      int row = idx >> 3;
      int c4  = (idx & 7) << 2;
      int gm  = blockIdx.x * TILE_M + row;
      if (gm >= M) gm = M - 1;            // clamp: pad rows replicate last row
      float4 v = *(const float4*)(base + (unsigned)gm * 256u + kk + c4);
      unsigned short* d = &As[row * LDS_PITCH + c4];
      d[0] = f2bf(v.x); d[1] = f2bf(v.y); d[2] = f2bf(v.z); d[3] = f2bf(v.w);
    }
    __syncthreads();

    v16bf bfrag[2];
#pragma unroll
    for (int nj = 0; nj < 2; nj++) {
      const unsigned short* bp =
          Wt + (unsigned)(nBase + nj * 16 + l16) * (unsigned)Ktot + k0 + half * 8;
      Frag32 f; f.lo = *(const uint4*)bp; f.hi = *(const uint4*)(bp + 16);
      bfrag[nj] = __builtin_bit_cast(v16bf, f);
    }
#pragma unroll
    for (int mi = 0; mi < 4; mi++) {
      const unsigned short* ap = &As[(mLoc + mi * 16 + l16) * LDS_PITCH + half * 8];
      Frag32 f; f.lo = *(const uint4*)ap; f.hi = *(const uint4*)(ap + 16);
      v16bf afrag = __builtin_bit_cast(v16bf, f);
      acc[mi][0] = __builtin_amdgcn_wmma_f32_16x16x32_bf16(
          false, afrag, false, bfrag[0], (short)0, acc[mi][0], false, false);
      acc[mi][1] = __builtin_amdgcn_wmma_f32_16x16x32_bf16(
          false, afrag, false, bfrag[1], (short)0, acc[mi][1], false, false);
    }
  }

  // Epilogue: unconditional stores (out buffers padded to grid extent).
#pragma unroll
  for (int nj = 0; nj < 2; nj++) {
    int gn = nBase + nj * 16 + l16;
    float bv = bias[gn];
#pragma unroll
    for (int mi = 0; mi < 4; mi++) {
      unsigned rowBase = (unsigned)(mBase + mi * 16 + half * 8);
#pragma unroll
      for (int r = 0; r < 8; r++) {
        float v = acc[mi][nj][r] + bv;
        if (doRelu) v = fmaxf(v, 0.0f);
        out[(rowBase + r) * (unsigned)N + gn] = v;
      }
    }
  }
}

// ---------------- p-weighted mean pool per graph (batch sorted) ----------------
__global__ __launch_bounds__(256) void pool(const float* __restrict__ h,
                                            const float* __restrict__ p,
                                            const int* __restrict__ batch,
                                            float* __restrict__ gmean) {
  int g = blockIdx.x, t = threadIdx.x;
  int s = lower_bound_i(batch, N_NODES, g);
  int e = lower_bound_i(batch, N_NODES, g + 1);
  float acc = 0.f;
  for (int i = s; i < e; i++) acc += h[(unsigned)i * 256u + t] * p[i];
  gmean[(unsigned)g * 256u + t] = acc / fmaxf((float)(e - s), 1.0f);
}

__global__ void gather_root(const float* __restrict__ h, const int* __restrict__ root_ptr,
                            float* __restrict__ out) {
  int g = blockIdx.x, t = threadIdx.x;
  out[(unsigned)g * 256u + t] = h[(unsigned)root_ptr[g] * 256u + t];
}

// ---------------- launch ----------------
extern "C" void kernel_launch(void* const* d_in, const int* in_sizes, int n_in,
                              void* d_out, int out_size, void* d_ws, size_t ws_size,
                              hipStream_t stream) {
  const float* x    = (const float*)d_in[0];
  const int*   src  = (const int*)d_in[1];
  const int*   dst  = (const int*)d_in[2];
  const float* p    = (const float*)d_in[3];
  const int*   batch= (const int*)d_in[4];
  const int*   rootp= (const int*)d_in[5];
  const float* Wl1  = (const float*)d_in[7];
  const float* Wr1  = (const float*)d_in[8];
  const float* b1   = (const float*)d_in[9];
  const float* Wl2  = (const float*)d_in[10];
  const float* Wr2  = (const float*)d_in[11];
  const float* b2   = (const float*)d_in[12];
  const float* Wlin = (const float*)d_in[13];
  const float* blin = (const float*)d_in[14];

  char* ws = (char*)d_ws;
  size_t off = 0;
  auto alloc = [&](size_t bytes) -> char* {
    char* r = ws + off; off += (bytes + 255) & ~(size_t)255; return r;
  };
  int*            row_ptr = (int*)           alloc((N_NODES + 1) * sizeof(int));
  float*          inv_deg = (float*)         alloc(N_NODES * sizeof(float));
  unsigned short* Wt1     = (unsigned short*)alloc((size_t)256 * 512 * 2);
  unsigned short* Wt2     = (unsigned short*)alloc((size_t)256 * 512 * 2);
  unsigned short* Wt3     = (unsigned short*)alloc((size_t)128 * 512 * 2);
  float*          agg     = (float*)         alloc((size_t)M_PAD * 256 * 4);
  float*          h1      = (float*)         alloc((size_t)M_PAD * 256 * 4);
  float*          h2      = (float*)         alloc((size_t)M_PAD * 256 * 4);
  float*          h2root  = (float*)         alloc((size_t)N_GRAPHS * 256 * 4);
  float*          gmean   = (float*)         alloc((size_t)N_GRAPHS * 256 * 4);
  (void)ws_size; (void)in_sizes; (void)n_in; (void)out_size;

  // CSR + inv degree
  build_rowptr<<<(N_NODES + 256) / 256, 256, 0, stream>>>(dst, row_ptr, inv_deg);

  // bf16 transposed weights: Wt[n][0:256]=Wl^T, Wt[n][256:512]=Wr^T
  wconv<<<256, 256, 0, stream>>>(Wl1, Wt1, 256, 256, 512, 0);
  wconv<<<256, 256, 0, stream>>>(Wr1, Wt1, 256, 256, 512, 256);
  wconv<<<256, 256, 0, stream>>>(Wl2, Wt2, 256, 256, 512, 0);
  wconv<<<256, 256, 0, stream>>>(Wr2, Wt2, 256, 256, 512, 256);
  wconv<<<256, 256, 0, stream>>>(Wlin, Wt3, 512, 128, 512, 0);

  dim3 g1(M_PAD / TILE_M, 256 / 128);   // (391, 2)

  // layer 1: h1 = relu([agg|x] @ [Wl1;Wr1] + b1)
  aggregate<<<(N_NODES + 3) / 4, 256, 0, stream>>>(x, src, row_ptr, inv_deg, agg);
  gemm_dual_bf16<<<g1, 256, 0, stream>>>(agg, x, Wt1, b1, h1, N_NODES, 256, 512, 1);

  // layer 2: h2 = [agg2|h1] @ [Wl2;Wr2] + b2
  aggregate<<<(N_NODES + 3) / 4, 256, 0, stream>>>(h1, src, row_ptr, inv_deg, agg);
  gemm_dual_bf16<<<g1, 256, 0, stream>>>(agg, h1, Wt2, b2, h2, N_NODES, 256, 512, 0);

  // readout
  pool<<<N_GRAPHS, 256, 0, stream>>>(h2, p, batch, gmean);
  gather_root<<<N_GRAPHS, 256, 0, stream>>>(h2, rootp, h2root);

  // head: out = [h2[root] | gmean] @ Wlin + blin  (M=256=2 tiles, N=128=1 tile)
  dim3 g3(N_GRAPHS / TILE_M, 1);
  gemm_dual_bf16<<<g3, 256, 0, stream>>>(h2root, gmean, Wt3, blin, (float*)d_out,
                                         N_GRAPHS, 128, 512, 0);
}
